// EdgeSamplingGumbel_17755394801755
// MI455X (gfx1250) — matrix operand order repaired and checked by reading, not compiled, not run on hardware
//
#include <hip/hip_runtime.h>

typedef __attribute__((ext_vector_type(16))) __bf16 v16bf;
typedef __attribute__((ext_vector_type(8)))  float  v8f;
typedef __attribute__((ext_vector_type(4)))  float  v4f;

#define NROWS 8192
#define DDIM  256
#define KSEL  5
#define TPAD  65   // 64 + 1 padding -> conflict-free LDS banks

union Frag16 { v16bf v; unsigned u[8]; };

// ---------------------------------------------------------------------------
// Kernel A: Poincare projection.
//  - per-row norm (fp32), projection scale 1/(max(|x|,1)*1.01)
//  - writes projected x as bf16 (row-major) for the WMMA GEMM
//  - writes per-row ||x_proj||^2 and 1/(1 - ||x_proj||^2)  (divide-free hot loop)
//  - copies the ORIGINAL x into d_out[0 .. N*D) (reference returns input x)
// ---------------------------------------------------------------------------
__global__ void proj_kernel(const float* __restrict__ x,
                            unsigned short* __restrict__ xbf,
                            float* __restrict__ xn2,
                            float* __restrict__ omi,
                            float* __restrict__ xout) {
  __shared__ float red[256];
  const int row = blockIdx.x;
  const int t   = threadIdx.x;
  const float v = x[(size_t)row * DDIM + t];
  red[t] = v * v;
  __syncthreads();
  for (int s = 128; s > 0; s >>= 1) {
    if (t < s) red[t] += red[t + s];
    __syncthreads();
  }
  const float nrm2 = red[0];
  const float nrm  = sqrtf(nrm2);
  const float inv  = 1.0f / (fmaxf(nrm, 1.0f) * 1.01f);
  const float xp   = v * inv;

  // fp32 -> bf16 round-to-nearest-even
  unsigned bits = __float_as_uint(xp);
  bits = (bits + 0x7FFFu + ((bits >> 16) & 1u)) >> 16;
  xbf[(size_t)row * DDIM + t] = (unsigned short)bits;

  xout[(size_t)row * DDIM + t] = v;            // pass-through output x
  if (t == 0) {
    const float n2 = nrm2 * inv * inv;         // ||x_proj||^2, fp32
    xn2[row] = n2;
    omi[row] = 1.0f / (1.0f - n2);             // den >= ~3.9e-4, safe
  }
}

// ---------------------------------------------------------------------------
// Kernel B: fused pq GEMM (bf16 WMMA) + hyperbolic distance + gumbel + top-5.
// Block = 128 threads (4 waves), owns a 16-row stripe. Each wave computes one
// 16x16 tile per column-block iteration (4 waves -> 64 columns / iter).
// ---------------------------------------------------------------------------
__global__ __launch_bounds__(128)
void fused_kernel(const unsigned short* __restrict__ xbf,
                  const float* __restrict__ xn2,
                  const float* __restrict__ omi,
                  const float* __restrict__ q,
                  const float* __restrict__ temp,
                  float* __restrict__ out) {
  __shared__ float tile[16][TPAD];
  __shared__ float mv[16][8 * KSEL];
  __shared__ int   mi[16][8 * KSEL];

  const int tid     = threadIdx.x;
  const int wid     = tid >> 5;        // wave id 0..3
  const int lane    = tid & 31;
  const int h       = lane >> 4;       // lane half
  const int ln16    = lane & 15;
  const int rowbase = blockIdx.x * 16;

  // temperature: exp(clip(T,-5,5)), uniform scalar
  float te = temp[0];
  te = fminf(5.0f, fmaxf(-5.0f, te));
  const float expT = __expf(te);

  const unsigned* xb32 = (const unsigned*)xbf;

  // A fragments: rows [rowbase, rowbase+16), full K=256 -> 8 chunks of K=32.
  // 16-bit A layout: lane m = lane%16; VGPR j<4: K = 8h+2j; j>=4: K = 16+8h+2(j-4)
  Frag16 afr[8];
  {
    const int arow = rowbase + ln16;
    #pragma unroll
    for (int kc = 0; kc < 8; ++kc) {
      #pragma unroll
      for (int p = 0; p < 8; ++p) {
        const int koff = 32 * kc + ((p < 4) ? (8 * h + 2 * p)
                                            : (16 + 8 * h + 2 * (p - 4)));
        afr[kc].u[p] = xb32[((size_t)arow * DDIM + koff) >> 1];
      }
    }
  }

  // Row-side constants for this lane's 8 accumulator rows (C layout: M = 8h+r)
  float xnr[8], invr[8];
  #pragma unroll
  for (int r = 0; r < 8; ++r) {
    xnr[r]  = xn2[rowbase + 8 * h + r];
    invr[r] = omi[rowbase + 8 * h + r];
  }

  // per-thread partial top-5 (thread tid handles row tid>>3, column octet tid&7)
  float t0 = -3.4e38f, t1 = -3.4e38f, t2 = -3.4e38f, t3 = -3.4e38f, t4 = -3.4e38f;
  int   i0 = 0, i1 = 0, i2 = 0, i3 = 0, i4 = 0;
  const int srow = tid >> 3;
  const int ssub = tid & 7;

  for (int cb = 0; cb < NROWS / 64; ++cb) {
    const int colbase = cb * 64;

    // ---- pq tile via WMMA (bf16 in, f32 accumulate) ----
    v8f acc = {0.f, 0.f, 0.f, 0.f, 0.f, 0.f, 0.f, 0.f};
    const int bcol = colbase + wid * 16 + ln16;
    #pragma unroll
    for (int kc = 0; kc < 8; ++kc) {
      Frag16 bfr;
      #pragma unroll
      for (int p = 0; p < 8; ++p) {
        const int koff = 32 * kc + ((p < 4) ? (8 * h + 2 * p)
                                            : (16 + 8 * h + 2 * (p - 4)));   // 16-bit B layout
        bfr.u[p] = xb32[((size_t)bcol * DDIM + koff) >> 1];
      }
      acc = __builtin_amdgcn_wmma_f32_16x16x32_bf16(
          false, afr[kc].v, false, bfr.v, (short)0, acc, false, false);
    }

    // ---- hyperbolic distance (divide-free), scaled & negated, into LDS ----
    const float xnc  = xn2[bcol];
    const float invc = omi[bcol];
    #pragma unroll
    for (int r = 0; r < 8; ++r) {
      const float pq  = acc[r];
      const float num = xnr[r] + xnc - 2.0f * pq;
      float arg = fmaf(2.0f * num, invr[r] * invc, 1.000001f);
      arg = fmaxf(arg, 1.000001f);                       // bf16 rounding guard
      const float ac = __logf(arg + sqrtf(fmaf(arg, arg, -1.0f)));  // acosh
      tile[8 * h + r][wid * 16 + ln16] = -(ac * ac) * expT;
    }
    __syncthreads();

    // ---- gumbel: val = -d + log(-log(q)); 16B non-temporal q loads ----
    #pragma unroll
    for (int e = 0; e < 2; ++e) {
      const int idx4 = tid + e * 128;          // 0..255 float4 slots
      const int r    = idx4 >> 4;              // 16 float4 per row
      const int c4   = idx4 & 15;
      const v4f qv = __builtin_nontemporal_load(
          (const v4f*)(q + (size_t)(rowbase + r) * NROWS + colbase + c4 * 4));
      #pragma unroll
      for (int j = 0; j < 4; ++j)
        tile[r][c4 * 4 + j] += __logf(-__logf(qv[j]));
    }
    __syncthreads();

    // ---- streaming top-5: 8 threads per row, 8 columns each ----
    #pragma unroll
    for (int c = 0; c < 8; ++c) {
      const int cl   = ssub * 8 + c;
      const float nv = tile[srow][cl];
      const int   ni = colbase + cl;
      if (nv > t4) {
        if (nv > t0)      { t4=t3;i4=i3; t3=t2;i3=i2; t2=t1;i2=i1; t1=t0;i1=i0; t0=nv;i0=ni; }
        else if (nv > t1) { t4=t3;i4=i3; t3=t2;i3=i2; t2=t1;i2=i1; t1=nv;i1=ni; }
        else if (nv > t2) { t4=t3;i4=i3; t3=t2;i3=i2; t2=nv;i2=ni; }
        else if (nv > t3) { t4=t3;i4=i3; t3=nv;i3=ni; }
        else              { t4=nv;i4=ni; }
      }
    }
    __syncthreads();
  }

  // ---- merge 8 partial top-5s per row, write outputs ----
  mv[srow][ssub * KSEL + 0] = t0;  mi[srow][ssub * KSEL + 0] = i0;
  mv[srow][ssub * KSEL + 1] = t1;  mi[srow][ssub * KSEL + 1] = i1;
  mv[srow][ssub * KSEL + 2] = t2;  mi[srow][ssub * KSEL + 2] = i2;
  mv[srow][ssub * KSEL + 3] = t3;  mi[srow][ssub * KSEL + 3] = i3;
  mv[srow][ssub * KSEL + 4] = t4;  mi[srow][ssub * KSEL + 4] = i4;
  __syncthreads();

  if (tid < 16) {
    float f0 = -3.4e38f, f1 = -3.4e38f, f2 = -3.4e38f, f3 = -3.4e38f, f4 = -3.4e38f;
    int   j0 = 0, j1 = 0, j2 = 0, j3 = 0, j4 = 0;
    for (int j = 0; j < 8 * KSEL; ++j) {
      const float nv = mv[tid][j];
      const int   ni = mi[tid][j];
      if (nv > f4) {
        if (nv > f0)      { f4=f3;j4=j3; f3=f2;j3=j2; f2=f1;j2=j1; f1=f0;j1=j0; f0=nv;j0=ni; }
        else if (nv > f1) { f4=f3;j4=j3; f3=f2;j3=j2; f2=f1;j2=j1; f1=nv;j1=ni; }
        else if (nv > f2) { f4=f3;j4=j3; f3=f2;j3=j2; f2=nv;j2=ni; }
        else if (nv > f3) { f4=f3;j4=j3; f3=nv;j3=ni; }
        else              { f4=nv;j4=ni; }
      }
    }
    const int grow = rowbase + tid;
    const size_t xb = (size_t)NROWS * DDIM;              // 2097152
    const size_t nk = (size_t)NROWS * KSEL;              // 40960
    float* ei0 = out + xb;                               // edge_index row 0
    float* ei1 = out + xb + nk;                          // edge_index row 1
    float* ew  = out + xb + 2 * nk;                      // edge_weights
    const size_t o = (size_t)grow * KSEL;
    ei0[o + 0] = (float)j0; ei0[o + 1] = (float)j1; ei0[o + 2] = (float)j2;
    ei0[o + 3] = (float)j3; ei0[o + 4] = (float)j4;
    ei1[o + 0] = (float)grow; ei1[o + 1] = (float)grow; ei1[o + 2] = (float)grow;
    ei1[o + 3] = (float)grow; ei1[o + 4] = (float)grow;
    ew[o + 0] = f0; ew[o + 1] = f1; ew[o + 2] = f2; ew[o + 3] = f3; ew[o + 4] = f4;
  }
}

extern "C" void kernel_launch(void* const* d_in, const int* in_sizes, int n_in,
                              void* d_out, int out_size, void* d_ws, size_t ws_size,
                              hipStream_t stream) {
  const float* x    = (const float*)d_in[0];
  const float* q    = (const float*)d_in[1];
  const float* temp = (const float*)d_in[2];
  float* out        = (float*)d_out;

  // workspace: [bf16 x_proj : N*D*2][fp32 norms : N*4][fp32 1/(1-n2) : N*4]
  unsigned short* xbf = (unsigned short*)d_ws;
  float* xn2 = (float*)((char*)d_ws + (size_t)NROWS * DDIM * sizeof(unsigned short));
  float* omi = xn2 + NROWS;

  proj_kernel<<<NROWS, 256, 0, stream>>>(x, xbf, xn2, omi, out);
  fused_kernel<<<NROWS / 16, 128, 0, stream>>>(xbf, xn2, omi, q, temp, out);
}
